// generator_17300128268317
// MI455X (gfx1250) — compile-verified, matrix operand order checked by reading, not run on hardware
//
#include <hip/hip_runtime.h>

// ---------------------------------------------------------------------------
// Types for CDNA5 WMMA (wave32, 16x16x32 bf16 -> f32 accumulate)
// ---------------------------------------------------------------------------
typedef __bf16 bf16_t;
typedef __attribute__((ext_vector_type(16))) __bf16 v16bf;
typedef __attribute__((ext_vector_type(4)))  __bf16 v4bf;
typedef __attribute__((ext_vector_type(8)))  float  v8f;
typedef __attribute__((ext_vector_type(4)))  float  v4f;

#define DEV __device__ __forceinline__

DEV float  bf2f(bf16_t x) { return (float)x; }
DEV bf16_t f2bf(float x)  { return (bf16_t)x; }

// Fragment from a row-major matrix where each lane owns one row and reads 16
// contiguous-ish bf16 along K (two 16-byte chunks -> 2x global_load_b128).
// Per ISA 7.12.2 (16-bit A 16x32): lanes 0-15 hold row r0+lane, K={0..7,16..23};
// lanes 16-31 hold row r0+lane-16, K={8..15,24..31}.
// Used for A tiles (rows of the activation matrix) AND for B tiles when the
// weight matrix is stored transposed (N x K row-major): lane row = output col.
DEV v16bf load_frag_rowK(const bf16_t* __restrict__ P, int r0, int k0, int ld) {
  int lane = threadIdx.x & 31;
  int half = lane >> 4;
  int r    = lane & 15;
  const bf16_t* row = P + (size_t)(r0 + r) * ld + k0;
  v16bf f;
#pragma unroll
  for (int i = 0; i < 16; ++i) {
    int k = i + 8 * half + ((i >= 8) ? 8 : 0);
    f[i] = row[k];
  }
  return f;
}

DEV v8f wmma_bf16(v16bf a, v16bf b, v8f c) {
  // (neg_a, A, neg_b, B, c_mod, C, reuse_a, reuse_b)
  return __builtin_amdgcn_wmma_f32_16x16x32_bf16(false, a, false, b, (short)0, c,
                                                 false, false);
}

// ---------------------------------------------------------------------------
// Problem constants
// ---------------------------------------------------------------------------
#define BATCH   4
#define N_PAIR  96          // 8*12
#define NF      32
#define C_FEAT  514
#define C_PAD   544         // 514 padded to multiple of 32 (17 K-steps)
#define M_ROWS  (BATCH * N_PAIR)             // 384
#define M_BIG   (BATCH * N_PAIR * N_PAIR)    // 36864
#define N_OUT   (N_PAIR * NF)                // 3072

// ---------------------------------------------------------------------------
// conv1 (3->64) fp32 input, fused ReLU, bf16 output
// ---------------------------------------------------------------------------
__global__ void k_conv1(const float* __restrict__ in, const float* __restrict__ w,
                        const float* __restrict__ bias, bf16_t* __restrict__ out) {
  const int H = 64, W = 96, Ci = 3;
  int sp = blockIdx.x * blockDim.x + threadIdx.x;
  if (sp >= H * W) return;
  int y = sp / W, x = sp % W;
  int co = blockIdx.y;               // gridDim.y = 64
  int b  = blockIdx.z;
  float acc = bias[co];
  const float* wc  = w + (size_t)co * Ci * 9;
  const float* inb = in + (size_t)b * Ci * H * W;
  for (int ci = 0; ci < Ci; ++ci) {
    const float* ip = inb + (size_t)ci * H * W;
    const float* wp = wc + ci * 9;
#pragma unroll
    for (int ky = 0; ky < 3; ++ky) {
      int iy = y + ky - 1;
      if (iy < 0 || iy >= H) continue;
#pragma unroll
      for (int kx = 0; kx < 3; ++kx) {
        int ix = x + kx - 1;
        if (ix < 0 || ix >= W) continue;
        acc += ip[iy * W + ix] * wp[ky * 3 + kx];
      }
    }
  }
  out[((size_t)(b * 64 + co) * H + y) * W + x] = f2bf(fmaxf(acc, 0.f));
}

// Generic direct 3x3 SAME conv, bf16 in, f32 accumulate, ReLU, bf16 out.
// blockIdx.y = co (weights wave-uniform -> scalar loads), blockIdx.z = batch.
__global__ void k_conv_direct(const bf16_t* __restrict__ in, const float* __restrict__ w,
                              const float* __restrict__ bias, bf16_t* __restrict__ out,
                              int Ci, int H, int W) {
  int sp = blockIdx.x * blockDim.x + threadIdx.x;
  if (sp >= H * W) return;
  int y = sp / W, x = sp % W;
  int co = blockIdx.y;
  int b  = blockIdx.z;
  int Co = gridDim.y;
  float acc = bias[co];
  const float*  wc  = w + (size_t)co * Ci * 9;
  const bf16_t* inb = in + (size_t)b * Ci * H * W;
  for (int ci = 0; ci < Ci; ++ci) {
    const bf16_t* ip = inb + (size_t)ci * H * W;
    const float*  wp = wc + ci * 9;
#pragma unroll
    for (int ky = 0; ky < 3; ++ky) {
      int iy = y + ky - 1;
      if (iy < 0 || iy >= H) continue;
#pragma unroll
      for (int kx = 0; kx < 3; ++kx) {
        int ix = x + kx - 1;
        if (ix < 0 || ix >= W) continue;
        acc += bf2f(ip[iy * W + ix]) * wp[ky * 3 + kx];
      }
    }
  }
  out[((size_t)(b * Co + co) * H + y) * W + x] = f2bf(fmaxf(acc, 0.f));
}

// 2x2 max pool, bf16. Out dims (BC,H,W); in dims (BC,2H,2W).
__global__ void k_pool(const bf16_t* __restrict__ in, bf16_t* __restrict__ out,
                       int total, int H, int W) {
  int i = blockIdx.x * blockDim.x + threadIdx.x;
  if (i >= total) return;
  int x = i % W, y = (i / W) % H, bc = i / (W * H);
  const bf16_t* p = in + ((size_t)bc * 2 * H + 2 * y) * (2 * W) + 2 * x;
  float m = fmaxf(fmaxf(bf2f(p[0]), bf2f(p[1])),
                  fmaxf(bf2f(p[2 * W]), bf2f(p[2 * W + 1])));
  out[i] = f2bf(m);
}

// ---------------------------------------------------------------------------
// Build F (B, 96, C_PAD): channels 0..511 = phi_im, 512/513 = coords, rest = 0
// ---------------------------------------------------------------------------
__global__ void k_build_F(const bf16_t* __restrict__ e4, bf16_t* __restrict__ F) {
  int i = blockIdx.x * blockDim.x + threadIdx.x;
  if (i >= BATCH * N_PAIR * C_PAD) return;
  int c = i % C_PAD;
  int n = (i / C_PAD) % N_PAIR;
  int b = i / (C_PAD * N_PAIR);
  float v;
  if (c < 512)        v = bf2f(e4[(size_t)(b * 512 + c) * N_PAIR + n]);
  else if (c == 512)  v = -1.f + 2.f * (float)(n / 8) / 11.f;  // xg.reshape(8,12)
  else if (c == 513)  v = -1.f + 2.f * (float)(n % 8) / 7.f;   // yg.reshape(8,12)
  else                v = 0.f;
  F[i] = f2bf(v);
}

// f32 KxN -> bf16 *transposed* NxK (row-major, K zero-padded to Kdst).
// Transposed storage makes WMMA B-fragment loads contiguous per lane
// (2x global_load_b128 instead of 16x global_load_u16).
__global__ void k_cvt_wT(const float* __restrict__ src, bf16_t* __restrict__ dst,
                         int Ksrc, int Kdst, int N, int rowOff) {
  int i = blockIdx.x * blockDim.x + threadIdx.x;
  if (i >= N * Kdst) return;
  int k = i % Kdst, n = i / Kdst;
  dst[i] = f2bf((k < Ksrc) ? src[(size_t)(rowOff + k) * N + n] : 0.f);
}

// ws = sen_embed @ Cm + g1_b  (tiny: 4x256, K=384, fp32 for accuracy)
__global__ void k_ws(const float* __restrict__ sen, const float* __restrict__ g1w,
                     const float* __restrict__ g1b, float* __restrict__ wsv) {
  int i = blockIdx.x * blockDim.x + threadIdx.x;
  if (i >= BATCH * 256) return;
  int n = i % 256, b = i / 256;
  float acc = g1b[n];
  for (int k = 0; k < 384; ++k)
    acc += sen[b * 384 + k] * g1w[(size_t)(1028 + k) * 256 + n];
  wsv[i] = acc;
}

// ---------------------------------------------------------------------------
// u = F@A, v = F@B   (WMMA bf16, M=384, N=256, K=544; f32 out)
// blockDim=128 (4 waves), one 16x16 tile per wave. 768 waves total.
// ---------------------------------------------------------------------------
__global__ void k_gemm_uv(const bf16_t* __restrict__ F, const bf16_t* __restrict__ wAt,
                          const bf16_t* __restrict__ wBt, float* __restrict__ u,
                          float* __restrict__ v) {
  int wave  = blockIdx.x * (blockDim.x >> 5) + (threadIdx.x >> 5);
  int which = wave / 384;
  int rem   = wave % 384;
  int m0 = (rem / 16) * 16;
  int n0 = (rem % 16) * 16;
  const bf16_t* Wt = which ? wBt : wAt;
  float*        O  = which ? v   : u;
  v8f acc = {};
  for (int k0 = 0; k0 < C_PAD; k0 += 32) {
    v16bf a = load_frag_rowK(F,  m0, k0, C_PAD);
    v16bf b = load_frag_rowK(Wt, n0, k0, C_PAD);   // transposed weights
    acc = wmma_bf16(a, b, acc);
  }
  int lane = threadIdx.x & 31, half = lane >> 4, n = lane & 15;
#pragma unroll
  for (int r = 0; r < 8; ++r)
    O[(size_t)(m0 + r + 8 * half) * 256 + n0 + n] = acc[r];
}

// h1[b,p,q,k] = relu(u[b,q,k] + v[b,p,k] + ws[b,k]) -> bf16, 4 elems/thread
__global__ void k_h1(const float* __restrict__ u, const float* __restrict__ v,
                     const float* __restrict__ wsv, bf16_t* __restrict__ h1) {
  size_t i = (size_t)blockIdx.x * blockDim.x + threadIdx.x;   // over M_BIG*64
  if (i >= (size_t)M_BIG * 64) return;
  int k4 = (int)(i % 64);                 // k = 4*k4
  int q  = (int)((i / 64) % N_PAIR);
  int p  = (int)((i / (64 * N_PAIR)) % N_PAIR);
  int b  = (int)(i / ((size_t)64 * N_PAIR * N_PAIR));
  v4f uu = *(const v4f*)(u   + (size_t)(b * N_PAIR + q) * 256 + 4 * k4);
  v4f vv = *(const v4f*)(v   + (size_t)(b * N_PAIR + p) * 256 + 4 * k4);
  v4f ww = *(const v4f*)(wsv + (size_t)b * 256 + 4 * k4);
  v4bf o;
#pragma unroll
  for (int j = 0; j < 4; ++j) o[j] = f2bf(fmaxf(uu[j] + vv[j] + ww[j], 0.f));
  *(v4bf*)(h1 + i * 4) = o;
}

// ---------------------------------------------------------------------------
// h2 = relu(h1 @ g2 + g2_b) -> bf16.  M=36864, N=256, K=256.
// Each wave owns one 16-col tile and 8 row tiles (128 rows); the 8 B fragments
// (full K) are loaded ONCE into 64 VGPRs and reused -> 8x less weight traffic.
// 4608 waves.
// ---------------------------------------------------------------------------
__global__ void k_h2(const bf16_t* __restrict__ h1, const bf16_t* __restrict__ wG2t,
                     const float* __restrict__ g2b, bf16_t* __restrict__ h2) {
  int wave    = blockIdx.x * (blockDim.x >> 5) + (threadIdx.x >> 5);
  int colTile = wave & 15;
  int rowGrp  = wave >> 4;            // 0..287, 128 rows each
  int n0 = colTile * 16;
  v16bf bfr[8];
#pragma unroll
  for (int kk = 0; kk < 8; ++kk) bfr[kk] = load_frag_rowK(wG2t, n0, kk * 32, 256);
  int lane = threadIdx.x & 31, half = lane >> 4, n = lane & 15;
  float bias = g2b[n0 + n];
  for (int t = 0; t < 8; ++t) {
    int m0 = rowGrp * 128 + t * 16;
    v8f acc = {};
#pragma unroll
    for (int kk = 0; kk < 8; ++kk) {
      v16bf a = load_frag_rowK(h1, m0, kk * 32, 256);
      acc = wmma_bf16(a, bfr[kk], acc);
    }
#pragma unroll
    for (int r = 0; r < 8; ++r)
      h2[(size_t)(m0 + r + 8 * half) * 256 + n0 + n] = f2bf(fmaxf(acc[r] + bias, 0.f));
  }
}

// ---------------------------------------------------------------------------
// Fused final GEMM + row-sum: phi[b,f] += sum_rows relu(h2_row @ g3[:,f] + g3_b[f])
// Never materializes the 452 MB `out` tensor.
// Grid: (192 col tiles, 36 row slabs of 1024 rows). 8 waves/block, 8 row
// tiles/wave; B fragments hoisted and reused across all 8 row tiles.
// Each slab lies within a single batch (1024 | 9216).
// ---------------------------------------------------------------------------
__global__ void k_out_sum(const bf16_t* __restrict__ h2, const bf16_t* __restrict__ wG3t,
                          const float* __restrict__ g3b, float* __restrict__ out) {
  __shared__ float red[8][32];
  int tid  = threadIdx.x;
  int wid  = tid >> 5, lane = tid & 31;
  int n0   = blockIdx.x * 16;
  int rowBase = blockIdx.y * 1024;
  int b    = rowBase / (N_PAIR * N_PAIR);
  int n    = lane & 15;
  v16bf bfr[8];
#pragma unroll
  for (int kk = 0; kk < 8; ++kk) bfr[kk] = load_frag_rowK(wG3t, n0, kk * 32, 256);
  float bias = g3b[n0 + n];
  float s = 0.f;
  for (int t = 0; t < 8; ++t) {
    int m0 = rowBase + (wid * 8 + t) * 16;
    if (t < 7) {  // prefetch next row tile's A rows (global_prefetch_b8)
      __builtin_prefetch(h2 + (size_t)(m0 + 16 + (lane & 15)) * 256, 0, 1);
    }
    v8f acc = {};
#pragma unroll
    for (int kk = 0; kk < 8; ++kk) {
      v16bf a = load_frag_rowK(h2, m0, kk * 32, 256);
      acc = wmma_bf16(a, bfr[kk], acc);
    }
#pragma unroll
    for (int r = 0; r < 8; ++r) s += fmaxf(acc[r] + bias, 0.f);
  }
  red[wid][lane] = s;
  __syncthreads();
  if (tid < 16) {
    float tot = 0.f;
#pragma unroll
    for (int w = 0; w < 8; ++w) tot += red[w][tid] + red[w][tid + 16];
    atomicAdd(&out[(size_t)b * N_OUT + n0 + tid], tot);
  }
}

__global__ void k_zero(float* __restrict__ p, int n) {
  int i = blockIdx.x * blockDim.x + threadIdx.x;
  if (i < n) p[i] = 0.f;
}

// ---------------------------------------------------------------------------
// Launch
// ---------------------------------------------------------------------------
extern "C" void kernel_launch(void* const* d_in, const int* in_sizes, int n_in,
                              void* d_out, int out_size, void* d_ws, size_t ws_size,
                              hipStream_t stream) {
  const float* x   = (const float*)d_in[0];
  const float* sen = (const float*)d_in[1];
  const float* c1w = (const float*)d_in[2];  const float* c1b = (const float*)d_in[3];
  const float* c2w = (const float*)d_in[4];  const float* c2b = (const float*)d_in[5];
  const float* c3w = (const float*)d_in[6];  const float* c3b = (const float*)d_in[7];
  const float* c4w = (const float*)d_in[8];  const float* c4b = (const float*)d_in[9];
  const float* g1w = (const float*)d_in[10]; const float* g1b = (const float*)d_in[11];
  const float* g2w = (const float*)d_in[12]; const float* g2b = (const float*)d_in[13];
  const float* g3w = (const float*)d_in[14]; const float* g3b = (const float*)d_in[15];
  float* out = (float*)d_out;

  char* wsp = (char*)d_ws;
  size_t off = 0;
  auto alloc = [&](size_t bytes) -> char* {
    char* p = wsp + off;
    off = (off + bytes + 255) & ~(size_t)255;
    return p;
  };
  bf16_t* e1   = (bf16_t*)alloc((size_t)4 * 64  * 64 * 96 * 2);
  bf16_t* p1   = (bf16_t*)alloc((size_t)4 * 64  * 32 * 48 * 2);
  bf16_t* e2   = (bf16_t*)alloc((size_t)4 * 128 * 32 * 48 * 2);
  bf16_t* p2   = (bf16_t*)alloc((size_t)4 * 128 * 16 * 24 * 2);
  bf16_t* e3   = (bf16_t*)alloc((size_t)4 * 256 * 16 * 24 * 2);
  bf16_t* p3   = (bf16_t*)alloc((size_t)4 * 256 * 8  * 12 * 2);
  bf16_t* e4   = (bf16_t*)alloc((size_t)4 * 512 * 8  * 12 * 2);
  bf16_t* F    = (bf16_t*)alloc((size_t)BATCH * N_PAIR * C_PAD * 2);
  bf16_t* wAt  = (bf16_t*)alloc((size_t)256 * C_PAD * 2);   // N x K (transposed)
  bf16_t* wBt  = (bf16_t*)alloc((size_t)256 * C_PAD * 2);
  bf16_t* wG2t = (bf16_t*)alloc((size_t)256 * 256 * 2);
  bf16_t* wG3t = (bf16_t*)alloc((size_t)N_OUT * 256 * 2);
  float*  u    = (float*)alloc((size_t)M_ROWS * 256 * 4);
  float*  v    = (float*)alloc((size_t)M_ROWS * 256 * 4);
  float*  wsv  = (float*)alloc((size_t)BATCH * 256 * 4);
  bf16_t* h1   = (bf16_t*)alloc((size_t)M_BIG * 256 * 2);
  bf16_t* h2   = (bf16_t*)alloc((size_t)M_BIG * 256 * 2);

  // Output accumulator must start at zero (d_out is poisoned).
  k_zero<<<(BATCH * N_OUT + 255) / 256, 256, 0, stream>>>(out, BATCH * N_OUT);

  // Encoder (L2-resident, 4% of FLOPs): direct conv + fused ReLU, bf16 tensors.
  k_conv1<<<dim3(24, 64, 4), 256, 0, stream>>>(x, c1w, c1b, e1);
  k_pool <<<(4 * 64 * 32 * 48 + 255) / 256, 256, 0, stream>>>(e1, p1, 4 * 64 * 32 * 48, 32, 48);
  k_conv_direct<<<dim3(6, 128, 4), 256, 0, stream>>>(p1, c2w, c2b, e2, 64, 32, 48);
  k_pool <<<(4 * 128 * 16 * 24 + 255) / 256, 256, 0, stream>>>(e2, p2, 4 * 128 * 16 * 24, 16, 24);
  k_conv_direct<<<dim3(2, 256, 4), 256, 0, stream>>>(p2, c3w, c3b, e3, 128, 16, 24);
  k_pool <<<(4 * 256 * 8 * 12 + 255) / 256, 256, 0, stream>>>(e3, p3, 4 * 256 * 8 * 12, 8, 12);
  k_conv_direct<<<dim3(1, 512, 4), 256, 0, stream>>>(p3, c4w, c4b, e4, 256, 8, 12);

  // Relation-network setup (weights converted to transposed NxK bf16).
  k_build_F<<<(BATCH * N_PAIR * C_PAD + 255) / 256, 256, 0, stream>>>(e4, F);
  k_cvt_wT<<<(256 * C_PAD + 255) / 256, 256, 0, stream>>>(g1w, wAt, C_FEAT, C_PAD, 256, 0);
  k_cvt_wT<<<(256 * C_PAD + 255) / 256, 256, 0, stream>>>(g1w, wBt, C_FEAT, C_PAD, 256, C_FEAT);
  k_cvt_wT<<<(256 * 256 + 255) / 256, 256, 0, stream>>>(g2w, wG2t, 256, 256, 256, 0);
  k_cvt_wT<<<(N_OUT * 256 + 255) / 256, 256, 0, stream>>>(g3w, wG3t, 256, 256, N_OUT, 0);
  k_ws<<<(BATCH * 256 + 255) / 256, 256, 0, stream>>>(sen, g1w, g1b, wsv);

  // WMMA GEMMs (96% of FLOPs).
  k_gemm_uv<<<192, 128, 0, stream>>>(F, wAt, wBt, u, v);                  // 768 waves
  k_h1<<<(int)(((size_t)M_BIG * 64 + 255) / 256), 256, 0, stream>>>(u, v, wsv, h1);
  k_h2<<<1152, 128, 0, stream>>>(h1, wG2t, g2b, h2);                      // 4608 waves
  k_out_sum<<<dim3(N_OUT / 16, M_BIG / 1024), 256, 0, stream>>>(h2, wG3t, g3b, out);
}
